// CapsNet_67053029425778
// MI455X (gfx1250) — matrix operand order repaired and checked by reading, not compiled, not run on hardware
//
#include <hip/hip_runtime.h>

typedef __attribute__((ext_vector_type(16))) _Float16 v16h;
typedef __attribute__((ext_vector_type(8)))  float    v8f;
typedef int v4i_vs __attribute__((vector_size(16)));   // matches builtin param pointee

union HFrag { v16h v; unsigned int u[8]; _Float16 h[16]; };

#define WMMA(a, b, c) __builtin_amdgcn_wmma_f32_16x16x32_f16( \
    false, (a), false, (b), (short)0, (c), false, false)

#if __has_builtin(__builtin_amdgcn_global_load_async_to_lds_b128)
#define HAVE_ASYNC_LDS 1
#define ASYNC_B128(gp, lp) __builtin_amdgcn_global_load_async_to_lds_b128(   \
    (__attribute__((address_space(1))) v4i_vs*)(const void*)(gp),            \
    (__attribute__((address_space(3))) v4i_vs*)(void*)(lp), 0, 0)
#else
#define HAVE_ASYNC_LDS 0
#endif

__device__ __forceinline__ void wait_async0() {
#if __has_builtin(__builtin_amdgcn_s_wait_asynccnt)
    __builtin_amdgcn_s_wait_asynccnt(0);
#else
    asm volatile("s_wait_asynccnt 0x0" ::: "memory");
#endif
}

// ---------------------------------------------------------------------------
// f32 -> f16 convert
// ---------------------------------------------------------------------------
__global__ void k_cvt(const float* __restrict__ in, _Float16* __restrict__ out, int n) {
    int i = blockIdx.x * 256 + threadIdx.x;
    if (i < n) out[i] = (_Float16)in[i];
}

// ---------------------------------------------------------------------------
// conv1 + relu: [512,1,28,28] -> [512,256,20,20], f16 output for WMMA consumer
// block = (b, group of 8 oc); x-plane staged in LDS
// ---------------------------------------------------------------------------
__global__ void __launch_bounds__(256) k_conv1(const float* __restrict__ x,
                                               const float* __restrict__ w,
                                               const float* __restrict__ bias,
                                               _Float16* __restrict__ h) {
    __shared__ float xs[784];
    int bx  = blockIdx.x;
    int b   = bx >> 5;
    int oc0 = (bx & 31) * 8;
    for (int i = threadIdx.x; i < 784; i += 256) xs[i] = x[(size_t)b * 784 + i];
    __syncthreads();
    for (int e = threadIdx.x; e < 8 * 400; e += 256) {
        int oc = oc0 + (e / 400);
        int s  = e % 400;
        int oy = s / 20, ox = s % 20;
        const float* wp = w + (size_t)oc * 81;
        float acc = bias[oc];
        #pragma unroll
        for (int ky = 0; ky < 9; ++ky)
            #pragma unroll
            for (int kx = 0; kx < 9; ++kx)
                acc += xs[(oy + ky) * 28 + ox + kx] * wp[ky * 9 + kx];
        acc = fmaxf(acc, 0.f);
        h[((size_t)b * 256 + oc) * 400 + s] = (_Float16)acc;
    }
}

// ---------------------------------------------------------------------------
// primary-caps conv as implicit GEMM with WMMA f32_16x16x32_f16.
// M = 256 (oc), K = 256*81 = 20736, N = 512*36 = 18432 (pos = b*36 + oy*6+ox).
// Block: 256 threads (8 waves) -> 128 oc x 64 pos tile; each wave owns one
// 16-oc sub-row and FOUR 16-pos D tiles (32 WMMA / block / k-chunk).
// A panel staged with async global->LDS b128 copies (ASYNCcnt), overlapped
// with the VALU im2col gather for B; (ic, k%81) tracked incrementally.
// ---------------------------------------------------------------------------
__global__ void __launch_bounds__(256) k_primconv(const _Float16* __restrict__ hf,
                                                  const _Float16* __restrict__ wf,
                                                  const float* __restrict__ bias,
                                                  float* __restrict__ p) {
    __shared__ __align__(16) unsigned int ldsA[2048];   // 128 oc x 16 dwords (32 f16)
    __shared__ __align__(16) _Float16 ldsB[2048];       // 64 pos x 32 k
    const unsigned int* ldsBu = (const unsigned int*)ldsB;

    int bx   = blockIdx.x;
    int ocg  = bx / 288;
    int posg = bx - ocg * 288;
    int oc0  = ocg * 128;
    int pos0 = posg * 64;

    int tid  = threadIdx.x;
    int wv   = tid >> 5;
    int lane = tid & 31;
    int half = lane >> 4;
    int m    = lane & 15;

    // hoisted im2col state: e = tid + q*256 -> pos_l = e>>5, kk = e&31
    int posoff[8], icq[8], rq[8];
    #pragma unroll
    for (int q = 0; q < 8; ++q) {
        int e     = tid + q * 256;
        int pos_l = e >> 5;
        int kk    = e & 31;
        int pos = pos0 + pos_l;
        int b   = pos / 36;
        int s   = pos - b * 36;
        int oy  = s / 6;
        int ox  = s - oy * 6;
        posoff[q] = b * 102400 + oy * 40 + ox * 2;  // base into hf[b][.][2oy][2ox]
        icq[q]    = 0;                               // ic = k/81 (k starts at kk<32)
        rq[q]     = kk;                              // r  = k%81
    }

    v8f accv[4] = {};

    for (int kc = 0; kc < 648; ++kc) {
        int k0 = kc * 32;
        // ---- stage A (contiguous): 512 x b128, async global->LDS ----
        #pragma unroll
        for (int q = 0; q < 2; ++q) {
            int id  = tid + q * 256;
            int row = id >> 2;
            int c8  = id & 3;
            const _Float16* gp = wf + (size_t)(oc0 + row) * 20736 + k0 + c8 * 8;
#if HAVE_ASYNC_LDS
            ASYNC_B128(gp, &ldsA[id * 4]);
#else
            *(uint4*)&ldsA[id * 4] = *(const uint4*)gp;
#endif
        }
        // ---- stage B: im2col gather [pos][k] (overlaps the async copies) ----
        #pragma unroll
        for (int q = 0; q < 8; ++q) {
            int ky = rq[q] / 9;
            int kx = rq[q] - ky * 9;
            ldsB[tid + q * 256] = hf[(size_t)posoff[q] + icq[q] * 400 + ky * 20 + kx];
            rq[q] += 32;
            if (rq[q] >= 81) { rq[q] -= 81; ++icq[q]; }
        }
#if HAVE_ASYNC_LDS
        wait_async0();
#endif
        __syncthreads();
        // ---- fragments + 4 WMMAs per wave ----
        HFrag a;
        int row = wv * 16 + m;
        #pragma unroll
        for (int jj = 0; jj < 8; ++jj) {
            // 16-bit A 16x32 layout: pairs K = {2jj | 16+2(jj-4)} + 8*half
            int ke = (jj < 4 ? jj * 2 : 16 + (jj - 4) * 2) + half * 8;
            a.u[jj] = ldsA[row * 16 + (ke >> 1)];
        }
        #pragma unroll
        for (int ps = 0; ps < 4; ++ps) {
            HFrag bb;
            #pragma unroll
            for (int jj = 0; jj < 8; ++jj) {
                // 16-bit B 32x16 layout: lane = N, K = {2jj | 16+2(jj-4)} + 8*half (+1)
                bb.u[jj] = ldsBu[(ps * 16 + m) * 16 + (jj < 4 ? jj : 8 + (jj - 4)) + half * 4];
            }
            accv[ps] = WMMA(a.v, bb.v, accv[ps]);
        }
        __syncthreads();
    }
    // D layout: VGPR r -> M = r + 8*half, N = lane&15
    #pragma unroll
    for (int r = 0; r < 8; ++r) {
        int oc   = oc0 + wv * 16 + half * 8 + r;
        float bv = bias[oc];
        #pragma unroll
        for (int ps = 0; ps < 4; ++ps)
            p[(size_t)oc * 18432 + pos0 + ps * 16 + m] = accv[ps][r] + bv;
    }
}

// ---------------------------------------------------------------------------
// squash primary caps: p[oc][b*36+s] -> u[b][c][8], c = (oc%32)*36 + s, d = oc/32
// ---------------------------------------------------------------------------
__global__ void k_squash(const float* __restrict__ p, float* __restrict__ u) {
    int t = blockIdx.x * 256 + threadIdx.x;
    if (t >= 512 * 1152) return;
    int b   = t / 1152;
    int c   = t - b * 1152;
    int col = c / 36;
    int s   = c - col * 36;
    float vals[8];
    float sq = 0.f;
    #pragma unroll
    for (int d = 0; d < 8; ++d) {
        float v = p[(size_t)(d * 32 + col) * 18432 + b * 36 + s];
        vals[d] = v;
        sq += v * v;
    }
    float scale = sq / ((1.f + sq) * sqrtf(sq) + 1e-30f);
    #pragma unroll
    for (int d = 0; d < 8; ++d) u[(size_t)t * 8 + d] = vals[d] * scale;
}

// ---------------------------------------------------------------------------
// fused priors + routing-by-agreement. One block per (o,b). Whole [1152,16]
// priors tile lives in LDS (72KB; CDNA5 WGP has 320KB) -> no HBM traffic for
// priors across the 3 iterations.
// ---------------------------------------------------------------------------
__global__ void __launch_bounds__(256) k_routing(const float* __restrict__ u,
                                                 const float* __restrict__ rw,
                                                 float* __restrict__ vcaps) {
    __shared__ float prior[1152 * 16];
    __shared__ float logits[1152];
    __shared__ float probs[1152];
    __shared__ float red[256];
    __shared__ float s_sh[16];
    __shared__ float v_sh[16];
    __shared__ float scal;

    int tid = threadIdx.x;
    int o   = blockIdx.x / 512;
    int b   = blockIdx.x - o * 512;

    // priors[c][d] = sum_i u[b,c,i] * rw[o,c,i,d]
    for (int e = tid; e < 1152 * 16; e += 256) {
        int c = e >> 4, d = e & 15;
        const float* up = u + ((size_t)b * 1152 + c) * 8;
        const float* wp = rw + (((size_t)o * 1152 + c) * 8) * 16 + d;
        float acc = 0.f;
        #pragma unroll
        for (int i = 0; i < 8; ++i) acc += up[i] * wp[i * 16];
        prior[e] = acc;
    }
    for (int c = tid; c < 1152; c += 256) logits[c] = 0.f;
    __syncthreads();

    for (int it = 0; it < 3; ++it) {
        // softmax over c (max-subtracted), keep unnormalized exp in probs
        float mx = -3.4e38f;
        for (int c = tid; c < 1152; c += 256) mx = fmaxf(mx, logits[c]);
        red[tid] = mx; __syncthreads();
        for (int off = 128; off > 0; off >>= 1) {
            if (tid < off) red[tid] = fmaxf(red[tid], red[tid + off]);
            __syncthreads();
        }
        float mmax = red[0]; __syncthreads();
        float se = 0.f;
        for (int c = tid; c < 1152; c += 256) {
            float ev = __expf(logits[c] - mmax);
            probs[c] = ev;
            se += ev;
        }
        red[tid] = se; __syncthreads();
        for (int off = 128; off > 0; off >>= 1) {
            if (tid < off) red[tid] += red[tid + off];
            __syncthreads();
        }
        float Z = red[0]; __syncthreads();

        // s[d] = (1/Z) * sum_c probs[c] * prior[c][d]
        int d = tid & 15, g = tid >> 4;
        float acc = 0.f;
        for (int c = g; c < 1152; c += 16) acc += probs[c] * prior[c * 16 + d];
        red[tid] = acc; __syncthreads();
        if (tid < 16) {
            float t2 = 0.f;
            for (int g2 = 0; g2 < 16; ++g2) t2 += red[g2 * 16 + tid];
            s_sh[tid] = t2 / Z;
        }
        __syncthreads();
        if (tid == 0) {
            float sq = 0.f;
            for (int d2 = 0; d2 < 16; ++d2) sq += s_sh[d2] * s_sh[d2];
            scal = sq / ((1.f + sq) * sqrtf(sq) + 1e-30f);
        }
        __syncthreads();
        if (tid < 16) v_sh[tid] = s_sh[tid] * scal;
        __syncthreads();
        if (it < 2) {
            for (int c = tid; c < 1152; c += 256) {
                float a2 = 0.f;
                #pragma unroll
                for (int d2 = 0; d2 < 16; ++d2) a2 += prior[c * 16 + d2] * v_sh[d2];
                logits[c] += a2;
            }
            __syncthreads();
        }
    }
    if (tid < 16) vcaps[((size_t)b * 10 + o) * 16 + tid] = v_sh[tid];
}

// ---------------------------------------------------------------------------
// lengths -> class softmax (to d_out) + argmax mask -> masked f16 [512,160]
// ---------------------------------------------------------------------------
__global__ void k_mask(const float* __restrict__ vcaps, float* __restrict__ classes,
                       _Float16* __restrict__ masked) {
    __shared__ float len[10];
    __shared__ float sm[2];
    __shared__ int arg;
    int b   = blockIdx.x;
    int tid = threadIdx.x;
    if (tid < 10) {
        const float* vp = vcaps + ((size_t)b * 10 + tid) * 16;
        float sq = 0.f;
        #pragma unroll
        for (int d = 0; d < 16; ++d) sq += vp[d] * vp[d];
        len[tid] = sqrtf(sq);
    }
    __syncthreads();
    if (tid == 0) {
        float mmax = len[0]; int am = 0;
        for (int o = 1; o < 10; ++o)
            if (len[o] > mmax) { mmax = len[o]; am = o; }
        float Z = 0.f;
        for (int o = 0; o < 10; ++o) Z += __expf(len[o] - mmax);
        sm[0] = mmax; sm[1] = Z; arg = am;
    }
    __syncthreads();
    if (tid < 10) classes[(size_t)b * 10 + tid] = __expf(len[tid] - sm[0]) / sm[1];
    for (int e = tid; e < 160; e += 32) {
        int o   = e >> 4;
        float v = (o == arg) ? vcaps[((size_t)b * 10 + o) * 16 + (e & 15)] : 0.f;
        masked[(size_t)b * 160 + e] = (_Float16)v;
    }
}

// ---------------------------------------------------------------------------
// generic FC via WMMA: out[n,j] = act( sum_k act_in[n,k]*w[j,k] + bias[j] )
// D tile = 16 j x 16 n, one tile per wave, fragments loaded straight from
// global (K contiguous for both operands). N = 512 fixed (32 n-tiles).
// mode 0: relu -> f16 ws ; mode 1: sigmoid -> f32 d_out.
// ---------------------------------------------------------------------------
__global__ void __launch_bounds__(256) k_fc_wmma(const _Float16* __restrict__ w,
                                                 const _Float16* __restrict__ act,
                                                 const float* __restrict__ bias,
                                                 int M, int K, int Mtiles, int mode,
                                                 _Float16* __restrict__ out16,
                                                 float* __restrict__ out32) {
    int tid  = threadIdx.x;
    int wv   = tid >> 5;
    int lane = tid & 31;
    int half = lane >> 4;
    int m    = lane & 15;
    int tt   = blockIdx.x * 8 + wv;          // wave-uniform
    if (tt >= Mtiles * 32) return;
    int jt = tt >> 5;
    int nt = tt & 31;
    v8f acc = {};
    int kch = K >> 5;
    const _Float16* arow = w   + (size_t)(jt * 16 + m) * K;
    const _Float16* brow = act + (size_t)(nt * 16 + m) * K;
    for (int kc = 0; kc < kch; ++kc) {
        int k0 = kc * 32;
        HFrag a, bb;
        #pragma unroll
        for (int jj = 0; jj < 8; ++jj) {
            int ke   = (jj < 4 ? jj * 2 : 16 + (jj - 4) * 2) + half * 8;
            a.u[jj]  = *(const unsigned int*)(arow + k0 + ke);
            bb.u[jj] = *(const unsigned int*)(brow + k0 + ke);
        }
        acc = WMMA(a.v, bb.v, acc);
    }
    #pragma unroll
    for (int r = 0; r < 8; ++r) {
        int j = jt * 16 + half * 8 + r;
        int n = nt * 16 + m;
        float v = acc[r] + bias[j];
        if (mode == 0) out16[(size_t)n * M + j] = (_Float16)fmaxf(v, 0.f);
        else           out32[(size_t)n * M + j] = 1.f / (1.f + __expf(-v));
    }
}

// ---------------------------------------------------------------------------
extern "C" void kernel_launch(void* const* d_in, const int* in_sizes, int n_in,
                              void* d_out, int out_size, void* d_ws, size_t ws_size,
                              hipStream_t stream) {
    const float* x   = (const float*)d_in[0];
    const float* c1w = (const float*)d_in[1];
    const float* c1b = (const float*)d_in[2];
    const float* pw  = (const float*)d_in[3];
    const float* pb  = (const float*)d_in[4];
    const float* rw  = (const float*)d_in[5];
    const float* f1w = (const float*)d_in[6];
    const float* f1b = (const float*)d_in[7];
    const float* f2w = (const float*)d_in[8];
    const float* f2b = (const float*)d_in[9];
    const float* f3w = (const float*)d_in[10];
    const float* f3b = (const float*)d_in[11];
    float* out = (float*)d_out;
    (void)in_sizes; (void)n_in; (void)out_size; (void)ws_size;

    char* base = (char*)d_ws;
    size_t off = 0;
    auto alloc = [&](size_t bytes) -> char* {
        off = (off + 255) & ~(size_t)255;
        char* ptr = base + off;
        off += bytes;
        return ptr;
    };
    _Float16* hf16   = (_Float16*)alloc((size_t)512 * 256 * 400 * 2);  // conv1 out
    _Float16* pw16   = (_Float16*)alloc((size_t)256 * 20736 * 2);      // prim_w f16
    float*    p      = (float*)   alloc((size_t)256 * 18432 * 4);      // prim conv out
    float*    u      = (float*)   alloc((size_t)512 * 1152 * 8 * 4);   // squashed caps
    float*    vcaps  = (float*)   alloc((size_t)512 * 10 * 16 * 4);    // routed caps
    _Float16* masked = (_Float16*)alloc((size_t)512 * 160 * 2);
    _Float16* f1w16  = (_Float16*)alloc((size_t)512 * 160 * 2);
    _Float16* f2w16  = (_Float16*)alloc((size_t)1024 * 512 * 2);
    _Float16* f3w16  = (_Float16*)alloc((size_t)784 * 1024 * 2);
    _Float16* h1     = (_Float16*)alloc((size_t)512 * 512 * 2);
    _Float16* h2     = (_Float16*)alloc((size_t)512 * 1024 * 2);

    int n;
    n = 256 * 20736; k_cvt<<<(n + 255) / 256, 256, 0, stream>>>(pw,  pw16,  n);
    n = 512 * 160;   k_cvt<<<(n + 255) / 256, 256, 0, stream>>>(f1w, f1w16, n);
    n = 1024 * 512;  k_cvt<<<(n + 255) / 256, 256, 0, stream>>>(f2w, f2w16, n);
    n = 784 * 1024;  k_cvt<<<(n + 255) / 256, 256, 0, stream>>>(f3w, f3w16, n);

    k_conv1<<<512 * 32, 256, 0, stream>>>(x, c1w, c1b, hf16);
    k_primconv<<<2 * 288, 256, 0, stream>>>(hf16, pw16, pb, p);
    n = 512 * 1152;  k_squash<<<(n + 255) / 256, 256, 0, stream>>>(p, u);
    k_routing<<<10 * 512, 256, 0, stream>>>(u, rw, vcaps);
    k_mask<<<512, 32, 0, stream>>>(vcaps, out, masked);
    k_fc_wmma<<<(32 * 32) / 8, 256, 0, stream>>>(f1w16, masked, f1b, 512, 160, 32, 0, h1, nullptr);
    k_fc_wmma<<<(64 * 32) / 8, 256, 0, stream>>>(f2w16, h1,     f2b, 1024, 512, 64, 0, h2, nullptr);
    k_fc_wmma<<<(49 * 32) / 8, 256, 0, stream>>>(f3w16, h2,     f3b, 784, 1024, 49, 1, nullptr, out + 5120);
}
